// CausalSelfAttention_57604101374209
// MI455X (gfx1250) — matrix operand order repaired and checked by reading, not compiled, not run on hardware
//
#include <hip/hip_runtime.h>
#include <cstdint>
#include <cstddef>

#define D_MODEL 1024
#define N_HEADS 16
#define HEAD_DIM 64
#define BATCH 2
#define SEQ 2048
#define ROWS (BATCH * SEQ) // 4096

typedef __bf16 v16bf __attribute__((ext_vector_type(16)));
typedef float  v8f   __attribute__((ext_vector_type(8)));
typedef unsigned int u32x4 __attribute__((ext_vector_type(4)));
typedef unsigned int u32x2 __attribute__((ext_vector_type(2)));
typedef float        f32x4 __attribute__((ext_vector_type(4)));

union Frag { u32x4 q[2]; v16bf v; };

__device__ __forceinline__ unsigned short f2bf(float f) {
  unsigned int u = __float_as_uint(f);
  u += 0x7FFFu + ((u >> 16) & 1u); // round-to-nearest-even
  return (unsigned short)(u >> 16);
}
__device__ __forceinline__ unsigned int pack2(float a, float b) {
  return (unsigned int)f2bf(a) | ((unsigned int)f2bf(b) << 16);
}

// CDNA5 async global->LDS copy (16B per lane), tracked by ASYNCcnt.
// lds = 32-bit LDS byte address (low 32 bits of generic shared pointer),
// g   = 64-bit global address.
__device__ __forceinline__ void async_copy16(unsigned lds, const void* g) {
  asm volatile("global_load_async_to_lds_b128 %0, %1, off"
               :: "v"(lds), "v"(g) : "memory");
}
__device__ __forceinline__ void wait_async() {
  asm volatile("s_wait_asynccnt 0" ::: "memory");
}

// ---------------------------------------------------------------------------
// Prep kernel 1: x (fp32) -> xb (bf16), flat copy-convert.
// ---------------------------------------------------------------------------
__global__ __launch_bounds__(256) void convert_x_kernel(
    const float* __restrict__ x, unsigned short* __restrict__ xb)
{
  size_t base = ((size_t)blockIdx.x * 256 + threadIdx.x) * 8;
  f32x4 f0 = *(const f32x4*)(x + base);
  f32x4 f1 = *(const f32x4*)(x + base + 4);
  u32x4 o;
  o[0] = pack2(f0[0], f0[1]); o[1] = pack2(f0[2], f0[3]);
  o[2] = pack2(f1[0], f1[1]); o[3] = pack2(f1[2], f1[3]);
  *(u32x4*)(xb + base) = o;
}

// ---------------------------------------------------------------------------
// Prep kernel 2: W[k][n] fp32 -> Wt[n][k] bf16 (LDS-tiled transpose so both
// global reads and writes are 16B coalesced). z selects Wq/Wk/Wv/Wo.
// ---------------------------------------------------------------------------
__global__ __launch_bounds__(256) void convert_w_kernel(
    const float* __restrict__ Wq, const float* __restrict__ Wk,
    const float* __restrict__ Wv, const float* __restrict__ Wo,
    unsigned short* __restrict__ Wqt, unsigned short* __restrict__ Wkt,
    unsigned short* __restrict__ Wvt, unsigned short* __restrict__ Wot)
{
  __shared__ unsigned char sT[64 * 144]; // 64 n-rows x 64 k bf16, 144B stride
  const int z = blockIdx.z;
  const float* W = (z == 0) ? Wq : (z == 1) ? Wk : (z == 2) ? Wv : Wo;
  unsigned short* Wt = (z == 0) ? Wqt : (z == 1) ? Wkt : (z == 2) ? Wvt : Wot;
  const int kBase = blockIdx.x * 64;
  const int nBase = blockIdx.y * 64;
  const int tid = threadIdx.x;

#pragma unroll
  for (int i = 0; i < 4; ++i) {           // read 64x64 fp32 tile
    int chunk = tid + i * 256;            // 1024 float4 chunks
    int r = chunk >> 4, c4 = chunk & 15;  // r = k row, c4*4 = n col
    f32x4 f = *(const f32x4*)(W + (size_t)(kBase + r) * D_MODEL + nBase + c4 * 4);
#pragma unroll
    for (int jn = 0; jn < 4; ++jn)
      *(unsigned short*)(sT + (c4 * 4 + jn) * 144 + r * 2) = f2bf(f[jn]);
  }
  __syncthreads();
#pragma unroll
  for (int i = 0; i < 2; ++i) {           // write 64x64 bf16, k-contiguous
    int chunk = tid + i * 256;            // 512 16B chunks
    int nl = chunk >> 3, k8 = chunk & 7;
    u32x4 d = *(const u32x4*)(sT + nl * 144 + k8 * 16);
    *(u32x4*)(Wt + (size_t)(nBase + nl) * D_MODEL + kBase + k8 * 8) = d;
  }
}

// ---------------------------------------------------------------------------
// Shared GEMM machinery: block tile 128(M) x 256(N), 8 waves of 64x64,
// K-step 32, double-buffered LDS, async global->LDS staging.
// A and Wt are both bf16; Wt is [N][K].
// ---------------------------------------------------------------------------
#define SA_BYTES (128 * 80)
#define SB_BYTES (256 * 80)
#define STAGE_BYTES (SA_BYTES + SB_BYTES)

__device__ __forceinline__ void stage_tiles(
    unsigned char* sBase,
    const unsigned short* __restrict__ A, const unsigned short* __restrict__ Wt,
    int mBase, int nBase, int kb, int tid)
{
  unsigned s32 = (unsigned)(uintptr_t)sBase; // LDS byte address (low 32 bits)
#pragma unroll
  for (int i = 0; i < 2; ++i) {          // A: 128x32 bf16 = 512 16B chunks
    int chunk = tid + i * 256;
    int r = chunk >> 2, c16 = chunk & 3;
    async_copy16(s32 + r * 80 + c16 * 16,
                 A + (size_t)(mBase + r) * D_MODEL + kb + c16 * 8);
  }
#pragma unroll
  for (int i = 0; i < 4; ++i) {          // B: 256 n-rows x 32 k bf16 = 1024 chunks
    int chunk = tid + i * 256;
    int n = chunk >> 2, c16 = chunk & 3;
    async_copy16(s32 + SA_BYTES + n * 80 + c16 * 16,
                 Wt + (size_t)(nBase + n) * D_MODEL + kb + c16 * 8);
  }
}

__device__ __forceinline__ void compute_step(
    const unsigned char* sBase, int mw, int nw, int lane, v8f acc[4][4])
{
  const unsigned char* sA = sBase;
  const unsigned char* sB = sBase + SA_BYTES;
  Frag a[4], b[4];
#pragma unroll
  for (int i = 0; i < 4; ++i) { // A frag: lane=row, K halves at (lane/16)*16+{0,32}
    const unsigned char* p = sA + (mw + i * 16 + (lane & 15)) * 80 + (lane >> 4) * 16;
    a[i].q[0] = *(const u32x4*)(p);
    a[i].q[1] = *(const u32x4*)(p + 32);
  }
#pragma unroll
  for (int j = 0; j < 4; ++j) { // B frag: lane=col, K halves at (lane/16)*16+{0,32}
    const unsigned char* p = sB + (nw + j * 16 + (lane & 15)) * 80 + (lane >> 4) * 16;
    b[j].q[0] = *(const u32x4*)(p);
    b[j].q[1] = *(const u32x4*)(p + 32);
  }
#pragma unroll
  for (int i = 0; i < 4; ++i)
#pragma unroll
    for (int j = 0; j < 4; ++j)
      acc[i][j] = __builtin_amdgcn_wmma_f32_16x16x32_bf16(
          false, a[i].v, false, b[j].v, (short)0, acc[i][j], false, false);
}

// ---------------------------------------------------------------------------
// Kernel: QKV projections. z: 0->Q (scaled by hd^-0.5), 1->K, 2->V transposed.
// ---------------------------------------------------------------------------
__global__ __launch_bounds__(256) void qkv_kernel(
    const unsigned short* __restrict__ xb,
    const unsigned short* __restrict__ Wqt, const float* __restrict__ bq,
    const unsigned short* __restrict__ Wkt, const float* __restrict__ bk,
    const unsigned short* __restrict__ Wvt, const float* __restrict__ bv,
    unsigned short* __restrict__ Q, unsigned short* __restrict__ K,
    unsigned short* __restrict__ Vt)
{
  __shared__ unsigned char smem[2 * STAGE_BYTES];

  const int z = blockIdx.z;
  const unsigned short* Wt = (z == 0) ? Wqt : (z == 1) ? Wkt : Wvt;
  const float* bias        = (z == 0) ? bq  : (z == 1) ? bk  : bv;

  const int nBase = blockIdx.x * 256;
  const int mBase = blockIdx.y * 128;
  const int tid  = threadIdx.x;
  const int lane = tid & 31;
  const int wave = tid >> 5;
  const int mw = (wave >> 2) * 64;
  const int nw = (wave & 3) * 64;

  v8f acc[4][4] = {};

  stage_tiles(smem, xb, Wt, mBase, nBase, 0, tid);
  wait_async();
  __syncthreads();
  for (int kk = 0; kk < D_MODEL / 32; ++kk) {
    int cur = kk & 1;
    if (kk + 1 < D_MODEL / 32)
      stage_tiles(smem + (cur ^ 1) * STAGE_BYTES, xb, Wt, mBase, nBase, (kk + 1) * 32, tid);
    compute_step(smem + cur * STAGE_BYTES, mw, nw, lane, acc);
    wait_async();
    __syncthreads();
  }

  // Uniform-mode epilogues (no per-element branching)
  if (z != 2) {
    unsigned short* o = (z == 0) ? Q : K;
    const float sc = (z == 0) ? 0.125f : 1.0f; // fold hd^-0.5 into Q
#pragma unroll
    for (int j = 0; j < 4; ++j) {
      int col = nBase + nw + j * 16 + (lane & 15);
      float bb = bias[col];
#pragma unroll
      for (int i = 0; i < 4; ++i)
#pragma unroll
        for (int g = 0; g < 8; ++g) {
          int row = mBase + mw + i * 16 + g + 8 * (lane >> 4);
          o[(size_t)row * D_MODEL + col] = f2bf((acc[i][j][g] + bb) * sc);
        }
    }
  } else {
#pragma unroll
    for (int j = 0; j < 4; ++j) {
      int col = nBase + nw + j * 16 + (lane & 15);
      float bb = bias[col];
#pragma unroll
      for (int i = 0; i < 4; ++i)
#pragma unroll
        for (int g = 0; g < 8; ++g) {
          int row = mBase + mw + i * 16 + g + 8 * (lane >> 4);
          int bi = row >> 11, t = row & (SEQ - 1);
          Vt[((size_t)(bi * D_MODEL + col)) * SEQ + t] = f2bf(acc[i][j][g] + bb);
        }
    }
  }
}

// ---------------------------------------------------------------------------
// Flash attention: one wave per (b, h, 32 Q rows). Q pre-scaled by 1/8.
// S = Q K^T (causal), online softmax, O = P V.  16 wmma per 32-col s-step.
// ---------------------------------------------------------------------------
__global__ __launch_bounds__(32) void attn_kernel(
    const unsigned short* __restrict__ Q,
    const unsigned short* __restrict__ K,
    const unsigned short* __restrict__ Vt,
    unsigned short* __restrict__ O)
{
  __shared__ unsigned char sP[2 * 16 * 80]; // two 16x32 bf16 P tiles
  const int lane = threadIdx.x & 31;
  const int blk  = blockIdx.x;
  const int tBlk = blk & (SEQ / 32 - 1);
  const int h    = (blk >> 6) & (N_HEADS - 1);
  const int b    = blk >> 10;
  const int tBase = tBlk * 32;

  // Q A-fragments: 2 row-tiles x 2 K=32 halves, loaded once
  Frag qa[2][2];
#pragma unroll
  for (int mi = 0; mi < 2; ++mi) {
    int t = tBase + mi * 16 + (lane & 15);
    const unsigned char* qrow =
        (const unsigned char*)(Q + (size_t)(b * SEQ + t) * D_MODEL + h * HEAD_DIM);
#pragma unroll
    for (int jj = 0; jj < 2; ++jj) {
      const unsigned char* p = qrow + jj * 64 + (lane >> 4) * 16;
      qa[mi][jj].q[0] = *(const u32x4*)p;
      qa[mi][jj].q[1] = *(const u32x4*)(p + 32);
    }
  }

  v8f o_acc[2][4] = {};
  float mrow[2][8], lrow[2][8];
#pragma unroll
  for (int mi = 0; mi < 2; ++mi)
#pragma unroll
    for (int g = 0; g < 8; ++g) { mrow[mi][g] = -3.0e38f; lrow[mi][g] = 0.f; }

  const int nSteps = tBlk + 1; // sBase = 0..tBase
  for (int ss = 0; ss < nSteps; ++ss) {
    const int sBase = ss * 32;

    // K^T B-fragments straight from row-major K (k = head dim contiguous)
    Frag kb_[2][2];
#pragma unroll
    for (int nt = 0; nt < 2; ++nt) {
      int s = sBase + nt * 16 + (lane & 15);
      const unsigned char* krow =
          (const unsigned char*)(K + (size_t)(b * SEQ + s) * D_MODEL + h * HEAD_DIM);
#pragma unroll
      for (int jj = 0; jj < 2; ++jj) {
        const unsigned char* p = krow + jj * 64 + (lane >> 4) * 16;
        kb_[nt][jj].q[0] = *(const u32x4*)p;
        kb_[nt][jj].q[1] = *(const u32x4*)(p + 32);
      }
    }

    v8f sacc[2][2] = {};
#pragma unroll
    for (int mi = 0; mi < 2; ++mi)
#pragma unroll
      for (int nt = 0; nt < 2; ++nt)
#pragma unroll
        for (int jj = 0; jj < 2; ++jj)
          sacc[mi][nt] = __builtin_amdgcn_wmma_f32_16x16x32_bf16(
              false, qa[mi][jj].v, false, kb_[nt][jj].v, (short)0, sacc[mi][nt], false, false);

    // Online softmax; P streamed to LDS (C layout -> A layout re-tiling)
#pragma unroll
    for (int mi = 0; mi < 2; ++mi) {
#pragma unroll
      for (int g = 0; g < 8; ++g) {
        int t = tBase + mi * 16 + g + 8 * (lane >> 4);
        float v0, v1;
        {
          int s0 = sBase + (lane & 15);
          int s1 = s0 + 16;
          v0 = (s0 > t) ? -3.0e38f : sacc[mi][0][g];
          v1 = (s1 > t) ? -3.0e38f : sacc[mi][1][g];
        }
        float m0 = fmaxf(v0, v1);
#pragma unroll
        for (int off = 1; off < 16; off <<= 1)
          m0 = fmaxf(m0, __shfl_xor(m0, off, 32));
        float mnew  = fmaxf(mrow[mi][g], m0);
        float scale = __expf(mrow[mi][g] - mnew);
        float p0 = __expf(v0 - mnew);
        float p1 = __expf(v1 - mnew);
        int m = g + 8 * (lane >> 4);
        *(unsigned short*)(sP + mi * 1280 + m * 80 + (lane & 15) * 2)      = f2bf(p0);
        *(unsigned short*)(sP + mi * 1280 + m * 80 + 32 + (lane & 15) * 2) = f2bf(p1);
        float rsum = p0 + p1;
#pragma unroll
        for (int off = 1; off < 16; off <<= 1)
          rsum += __shfl_xor(rsum, off, 32);
        lrow[mi][g] = lrow[mi][g] * scale + rsum;
        mrow[mi][g] = mnew;
#pragma unroll
        for (int jt = 0; jt < 4; ++jt) o_acc[mi][jt][g] *= scale;
      }
    }
    __syncthreads();
    Frag pa[2];
#pragma unroll
    for (int mi = 0; mi < 2; ++mi) {
      const unsigned char* p = sP + mi * 1280 + (lane & 15) * 80 + (lane >> 4) * 16;
      pa[mi].q[0] = *(const u32x4*)p;
      pa[mi].q[1] = *(const u32x4*)(p + 32);
    }
    __syncthreads();

    // O += P @ V; V fragment reused across both row-tiles
#pragma unroll
    for (int jt = 0; jt < 4; ++jt) {
      int j = jt * 16 + (lane & 15);
      const unsigned char* vrow =
          (const unsigned char*)(Vt + ((size_t)(b * D_MODEL + h * HEAD_DIM + j)) * SEQ + sBase);
      Frag vb;
      const unsigned char* p = vrow + (lane >> 4) * 16;
      vb.q[0] = *(const u32x4*)p;
      vb.q[1] = *(const u32x4*)(p + 32);
#pragma unroll
      for (int mi = 0; mi < 2; ++mi)
        o_acc[mi][jt] = __builtin_amdgcn_wmma_f32_16x16x32_bf16(
            false, pa[mi].v, false, vb.v, (short)0, o_acc[mi][jt], false, false);
    }
  }

  // Normalize and store O (bf16, row-major [4096,1024])
#pragma unroll
  for (int mi = 0; mi < 2; ++mi)
#pragma unroll
    for (int jt = 0; jt < 4; ++jt) {
      int col = h * HEAD_DIM + jt * 16 + (lane & 15);
#pragma unroll
      for (int g = 0; g < 8; ++g) {
        int row = b * SEQ + tBase + mi * 16 + g + 8 * (lane >> 4);
        O[(size_t)row * D_MODEL + col] = f2bf(o_acc[mi][jt][g] / lrow[mi][g]);
      }
    }
}

// ---------------------------------------------------------------------------
// Kernel: output projection. out = O @ Wo + bo, fp32 output.
// ---------------------------------------------------------------------------
__global__ __launch_bounds__(256) void proj_kernel(
    const unsigned short* __restrict__ A,   // attention output, bf16
    const unsigned short* __restrict__ Wt,  // Wo transposed bf16 [N][K]
    const float* __restrict__ bias,
    float* __restrict__ Cout)
{
  __shared__ unsigned char smem[2 * STAGE_BYTES];

  const int nBase = blockIdx.x * 256;
  const int mBase = blockIdx.y * 128;
  const int tid  = threadIdx.x;
  const int lane = tid & 31;
  const int wave = tid >> 5;
  const int mw = (wave >> 2) * 64;
  const int nw = (wave & 3) * 64;

  v8f acc[4][4] = {};

  stage_tiles(smem, A, Wt, mBase, nBase, 0, tid);
  wait_async();
  __syncthreads();
  for (int kk = 0; kk < D_MODEL / 32; ++kk) {
    int cur = kk & 1;
    if (kk + 1 < D_MODEL / 32)
      stage_tiles(smem + (cur ^ 1) * STAGE_BYTES, A, Wt, mBase, nBase, (kk + 1) * 32, tid);
    compute_step(smem + cur * STAGE_BYTES, mw, nw, lane, acc);
    wait_async();
    __syncthreads();
  }

#pragma unroll
  for (int j = 0; j < 4; ++j) {
    int col = nBase + nw + j * 16 + (lane & 15);
    float bb = bias[col];
#pragma unroll
    for (int i = 0; i < 4; ++i)
#pragma unroll
      for (int g = 0; g < 8; ++g) {
        int row = mBase + mw + i * 16 + g + 8 * (lane >> 4);
        Cout[(size_t)row * D_MODEL + col] = acc[i][j][g] + bb;
      }
  }
}

// ---------------------------------------------------------------------------
extern "C" void kernel_launch(void* const* d_in, const int* in_sizes, int n_in,
                              void* d_out, int out_size, void* d_ws, size_t ws_size,
                              hipStream_t stream) {
  const float* x  = (const float*)d_in[0];
  // d_in[1] = additive causal mask: applied analytically in attn_kernel
  const float* Wq = (const float*)d_in[2];
  const float* bq = (const float*)d_in[3];
  const float* Wk = (const float*)d_in[4];
  const float* bk = (const float*)d_in[5];
  const float* Wv = (const float*)d_in[6];
  const float* bv = (const float*)d_in[7];
  const float* Wo = (const float*)d_in[8];
  const float* bo = (const float*)d_in[9];
  float* out = (float*)d_out;

  // Workspace layout (all bf16):
  //  xb 8M | Wqt 2M | Wkt 2M | Wvt 2M | Wot 2M | Q 8M | K 8M | Vt 8M | O 8M
  char* ws = (char*)d_ws;
  const size_t MB = 1024 * 1024;
  unsigned short* xb  = (unsigned short*)(ws);
  unsigned short* Wqt = (unsigned short*)(ws + 8  * MB);
  unsigned short* Wkt = (unsigned short*)(ws + 10 * MB);
  unsigned short* Wvt = (unsigned short*)(ws + 12 * MB);
  unsigned short* Wot = (unsigned short*)(ws + 14 * MB);
  unsigned short* Q   = (unsigned short*)(ws + 16 * MB);
  unsigned short* K   = (unsigned short*)(ws + 24 * MB);
  unsigned short* Vt  = (unsigned short*)(ws + 32 * MB);
  unsigned short* O   = (unsigned short*)(ws + 40 * MB);

  convert_x_kernel<<<dim3((ROWS * D_MODEL) / (256 * 8)), 256, 0, stream>>>(x, xb);
  convert_w_kernel<<<dim3(D_MODEL / 64, D_MODEL / 64, 4), 256, 0, stream>>>(
      Wq, Wk, Wv, Wo, Wqt, Wkt, Wvt, Wot);

  dim3 gQkv(D_MODEL / 256, ROWS / 128, 3); // (4, 32, 3)
  qkv_kernel<<<gQkv, 256, 0, stream>>>(xb, Wqt, bq, Wkt, bk, Wvt, bv, Q, K, Vt);

  dim3 gAttn(BATCH * N_HEADS * (SEQ / 32)); // 2048 waves
  attn_kernel<<<gAttn, 32, 0, stream>>>(Q, K, Vt, O);

  dim3 gProj(D_MODEL / 256, ROWS / 128); // (4, 32)
  proj_kernel<<<gProj, 256, 0, stream>>>(O, Wot, bo, out);
}
